// SequentialLlama4TextMoe_68874095558952
// MI455X (gfx1250) — compile-verified, matrix operand order checked by reading, not run on hardware
//
#include <hip/hip_runtime.h>
#include <stdint.h>

#define T_TOK 8192
#define HDIM  1024
#define IDIM  2048
#define NEXP  16

#define KSTEP  32
#define TILE_M 64
#define TILE_N 128
#define LDSS   (KSTEP + 8)   // 40 elems = 80B row stride (multiple of 16B)

typedef __attribute__((ext_vector_type(16))) __bf16 v16bf;
typedef __attribute__((ext_vector_type(8)))  __bf16 v8bf;
typedef __attribute__((ext_vector_type(8)))  float  v8f;

// ---- fragment loaders: one lane's 16 bf16 elements of a 16x32/32x16 operand.
// K chunks per CDNA5 16-bit layout: {8g..8g+7} and {16+8g..16+8g+7}, g=lane>>4.
__device__ __forceinline__ v16bf load_frag(const __bf16* __restrict__ p, int g) {
  v8bf b0 = *(const v8bf*)(p + 8 * g);        // 16B load
  v8bf b1 = *(const v8bf*)(p + 16 + 8 * g);   // 16B load
  return __builtin_shufflevector(b0, b1, 0, 1, 2, 3, 4, 5, 6, 7,
                                         8, 9, 10, 11, 12, 13, 14, 15);
}
__device__ __forceinline__ v16bf load_frag(const float* __restrict__ p, int g) {
  v8f c0 = *(const v8f*)(p + 8 * g);          // 32B load
  v8f c1 = *(const v8f*)(p + 16 + 8 * g);
  v8bf b0 = __builtin_convertvector(c0, v8bf);  // native v_cvt (RNE)
  v8bf b1 = __builtin_convertvector(c1, v8bf);
  return __builtin_shufflevector(b0, b1, 0, 1, 2, 3, 4, 5, 6, 7,
                                         8, 9, 10, 11, 12, 13, 14, 15);
}

__device__ __forceinline__ v8f wmma_bf16(v16bf a, v16bf b, v8f c) {
  return __builtin_amdgcn_wmma_f32_16x16x32_bf16(false, a, false, b,
                                                 (short)0, c, false, false);
}

// ---------------------------------------------------------------- small utils
__global__ void init_kernel(int* counts, int* cnt2, int* sh_off, int* sh_cnt,
                            int* idperm) {
  const int t = blockIdx.x * 256 + threadIdx.x;
  if (t < NEXP) { counts[t] = 0; cnt2[t] = 0; }
  if (t == 0)   { sh_off[0] = 0; sh_cnt[0] = T_TOK; }
  if (t < T_TOK) idperm[t] = t;
}

// fp32 -> bf16, 8 elements/thread (sizes are multiples of 2048)
__global__ __launch_bounds__(256) void cvtw_kernel(const float* __restrict__ src,
                                                   __bf16* __restrict__ dst) {
  const size_t i = ((size_t)blockIdx.x * 256 + threadIdx.x) * 8;
  v8f f = *(const v8f*)(src + i);
  *(v8bf*)(dst + i) = __builtin_convertvector(f, v8bf);
}

// one wave per token: lane = (expert, half-of-H); top-1 + sigmoid score
__global__ __launch_bounds__(256) void router_kernel(
    const float* __restrict__ x, const float* __restrict__ rw,
    float* __restrict__ logits, float* __restrict__ score,
    int* __restrict__ eid, int* __restrict__ counts) {
  const int lane = threadIdx.x & 31, wave = threadIdx.x >> 5;
  const int t = blockIdx.x * 8 + wave;
  if (t >= T_TOK) return;
  const int e = lane & 15, h = lane >> 4;
  const float4* xr = (const float4*)(x + (size_t)t * HDIM + h * 512);
  const float4* wr = (const float4*)(rw + (size_t)e * HDIM + h * 512);
  float acc = 0.f;
  for (int i = 0; i < 128; ++i) {
    float4 a = xr[i], b = wr[i];
    acc += a.x * b.x + a.y * b.y + a.z * b.z + a.w * b.w;
  }
  acc += __shfl_xor(acc, 16);
  if (lane < 16) logits[(size_t)t * NEXP + e] = acc;
  float bv = acc; int bi = e;
#pragma unroll
  for (int off = 8; off; off >>= 1) {
    float ov = __shfl_xor(bv, off);
    int   oi = __shfl_xor(bi, off);
    if (ov > bv || (ov == bv && oi < bi)) { bv = ov; bi = oi; }
  }
  if (lane == 0) {
    score[t] = 1.0f / (1.0f + __expf(-bv));
    eid[t] = bi;
    atomicAdd(&counts[bi], 1);
  }
}

__global__ void scan_kernel(const int* __restrict__ counts, int* __restrict__ offs) {
  if (threadIdx.x == 0) {
    int s = 0;
    for (int e = 0; e < NEXP; ++e) { offs[e] = s; s += counts[e]; }
  }
}

__global__ void scatter_kernel(const int* __restrict__ eid,
                               const float* __restrict__ score,
                               const int* __restrict__ offs, int* cnt2,
                               int* __restrict__ gperm, float* __restrict__ gscore) {
  const int t = blockIdx.x * 256 + threadIdx.x;
  if (t >= T_TOK) return;
  const int e = eid[t];
  const int p = offs[e] + atomicAdd(&cnt2[e], 1);
  gperm[p] = t;
  gscore[p] = score[t];
}

// -------------------------------------------------- WMMA GEMM 1: gate/up+silu
// hbuf[pos, :] = silu(X@gw^T) * (X@up^T), X rows gathered via perm.
// Block tile M64 x N128; 8 waves as 2(M) x 4(N); wave tile M32 x N32.
template <typename WT>
__global__ __launch_bounds__(256) void gateup_kernel(
    const __bf16* __restrict__ xb,
    const WT* __restrict__ gw_all, const WT* __restrict__ uw_all,
    __bf16* __restrict__ hbuf,
    const int* __restrict__ perm, const int* __restrict__ seg_off,
    const int* __restrict__ seg_cnt) {
  __shared__ __bf16 xs[2][TILE_M][LDSS];
  const int seg = blockIdx.z;
  const int cnt = seg_cnt[seg];
  const int m0  = blockIdx.x * TILE_M;
  if (m0 >= cnt) return;
  const int base = seg_off[seg];
  const int tid = threadIdx.x;
  const int lane = tid & 31, wave = tid >> 5;
  const int waveM = wave >> 2, waveN = wave & 3;   // 2 x 4 wave grid
  const int laneN = lane & 15, g = lane >> 4;
  const int nWave = blockIdx.y * TILE_N + waveN * 32;

  const WT* gw = gw_all + (size_t)seg * IDIM * HDIM;
  const WT* uw = uw_all + (size_t)seg * IDIM * HDIM;

  const int srow = tid >> 2;            // 0..63
  const int scol = (tid & 3) * 8;       // 0,8,16,24
  int sl = m0 + srow; if (sl >= cnt) sl = cnt - 1;
  const __bf16* xrow = xb + (size_t)perm[base + sl] * HDIM;

  const v8f vzero = {0.f, 0.f, 0.f, 0.f, 0.f, 0.f, 0.f, 0.f};
  v8f gacc[2][2], uacc[2][2];
#pragma unroll
  for (int mi = 0; mi < 2; ++mi)
#pragma unroll
    for (int nt = 0; nt < 2; ++nt) { gacc[mi][nt] = vzero; uacc[mi][nt] = vzero; }

  *(v8bf*)&xs[0][srow][scol] = *(const v8bf*)&xrow[scol];
  int buf = 0;
  for (int kb = 0; kb < HDIM; kb += KSTEP) {
    __syncthreads();
    if (kb + KSTEP < HDIM)
      *(v8bf*)&xs[buf ^ 1][srow][scol] = *(const v8bf*)&xrow[kb + KSTEP + scol];

    const v16bf a0 = load_frag(&xs[buf][waveM * 32 + laneN][0], g);
    const v16bf a1 = load_frag(&xs[buf][waveM * 32 + 16 + laneN][0], g);

#pragma unroll
    for (int nt = 0; nt < 2; ++nt) {
      const size_t nrow = (size_t)(nWave + nt * 16 + laneN);
      const v16bf bg = load_frag(gw + nrow * HDIM + kb, g);
      gacc[0][nt] = wmma_bf16(a0, bg, gacc[0][nt]);
      gacc[1][nt] = wmma_bf16(a1, bg, gacc[1][nt]);
      const v16bf bu = load_frag(uw + nrow * HDIM + kb, g);
      uacc[0][nt] = wmma_bf16(a0, bu, uacc[0][nt]);
      uacc[1][nt] = wmma_bf16(a1, bu, uacc[1][nt]);
    }
    buf ^= 1;
  }

#pragma unroll
  for (int mi = 0; mi < 2; ++mi)
#pragma unroll
    for (int v = 0; v < 8; ++v) {
      const int rowSeg = m0 + waveM * 32 + mi * 16 + v + 8 * g;  // D row=v+8g
      if (rowSeg >= cnt) continue;
      __bf16* hr = hbuf + (size_t)(base + rowSeg) * IDIM;
#pragma unroll
      for (int nt = 0; nt < 2; ++nt) {
        float gv = gacc[mi][nt][v], uv = uacc[mi][nt][v];
        float hv = (gv / (1.0f + __expf(-gv))) * uv;   // silu(g)*u
        hr[nWave + nt * 16 + laneN] = (__bf16)hv;
      }
    }
}

// -------------------------------------------------------- WMMA GEMM 2: down
// out[perm[pos], :] (=|+=) (hbuf[pos,:] @ dw^T) * score[pos]
template <typename WT>
__global__ __launch_bounds__(256) void down_kernel(
    const __bf16* __restrict__ hbuf,
    const WT* __restrict__ dw_all, float* __restrict__ out,
    const int* __restrict__ perm, const float* __restrict__ gscore,
    const int* __restrict__ seg_off, const int* __restrict__ seg_cnt,
    int accumulate) {
  __shared__ __bf16 hsh[2][TILE_M][LDSS];
  const int seg = blockIdx.z;
  const int cnt = seg_cnt[seg];
  const int m0  = blockIdx.x * TILE_M;
  if (m0 >= cnt) return;
  const int base = seg_off[seg];
  const int tid = threadIdx.x;
  const int lane = tid & 31, wave = tid >> 5;
  const int waveM = wave >> 2, waveN = wave & 3;
  const int laneN = lane & 15, g = lane >> 4;
  const int nWave = blockIdx.y * TILE_N + waveN * 32;

  const WT* dw = dw_all + (size_t)seg * HDIM * IDIM;

  const int srow = tid >> 2;
  const int scol = (tid & 3) * 8;
  int sl = m0 + srow; if (sl >= cnt) sl = cnt - 1;
  const __bf16* hrow = hbuf + (size_t)(base + sl) * IDIM;

  const v8f vzero = {0.f, 0.f, 0.f, 0.f, 0.f, 0.f, 0.f, 0.f};
  v8f acc[2][2];
#pragma unroll
  for (int mi = 0; mi < 2; ++mi)
#pragma unroll
    for (int nt = 0; nt < 2; ++nt) acc[mi][nt] = vzero;

  *(v8bf*)&hsh[0][srow][scol] = *(const v8bf*)&hrow[scol];
  int buf = 0;
  for (int kb = 0; kb < IDIM; kb += KSTEP) {
    __syncthreads();
    if (kb + KSTEP < IDIM)
      *(v8bf*)&hsh[buf ^ 1][srow][scol] = *(const v8bf*)&hrow[kb + KSTEP + scol];

    const v16bf a0 = load_frag(&hsh[buf][waveM * 32 + laneN][0], g);
    const v16bf a1 = load_frag(&hsh[buf][waveM * 32 + 16 + laneN][0], g);

#pragma unroll
    for (int nt = 0; nt < 2; ++nt) {
      const size_t nrow = (size_t)(nWave + nt * 16 + laneN);
      const v16bf bd = load_frag(dw + nrow * IDIM + kb, g);
      acc[0][nt] = wmma_bf16(a0, bd, acc[0][nt]);
      acc[1][nt] = wmma_bf16(a1, bd, acc[1][nt]);
    }
    buf ^= 1;
  }

  int toks[2][8]; float sc[2][8];
#pragma unroll
  for (int mi = 0; mi < 2; ++mi)
#pragma unroll
    for (int v = 0; v < 8; ++v) {
      int rowSeg = m0 + waveM * 32 + mi * 16 + v + 8 * g;
      int rs = rowSeg < cnt ? rowSeg : cnt - 1;
      toks[mi][v] = perm[base + rs];
      sc[mi][v]   = gscore ? gscore[base + rs] : 1.0f;
    }
#pragma unroll
  for (int mi = 0; mi < 2; ++mi)
#pragma unroll
    for (int v = 0; v < 8; ++v) {
      if (m0 + waveM * 32 + mi * 16 + v + 8 * g >= cnt) continue;
#pragma unroll
      for (int nt = 0; nt < 2; ++nt) {
        const size_t o = (size_t)toks[mi][v] * HDIM + nWave + nt * 16 + laneN;
        const float val = acc[mi][nt][v] * sc[mi][v];
        out[o] = accumulate ? out[o] + val : val;
      }
    }
}

// ---------------------------------------------------------------------- host
extern "C" void kernel_launch(void* const* d_in, const int* in_sizes, int n_in,
                              void* d_out, int out_size, void* d_ws, size_t ws_size,
                              hipStream_t stream) {
  (void)in_sizes; (void)n_in; (void)out_size;
  const float* x   = (const float*)d_in[0];
  const float* rw  = (const float*)d_in[1];
  const float* sgw = (const float*)d_in[2];
  const float* suw = (const float*)d_in[3];
  const float* sdw = (const float*)d_in[4];
  const float* egw = (const float*)d_in[5];
  const float* euw = (const float*)d_in[6];
  const float* edw = (const float*)d_in[7];

  float* out    = (float*)d_out;                 // [T, H]
  float* logits = out + (size_t)T_TOK * HDIM;    // [T, E]

  uint8_t* w = (uint8_t*)d_ws;
  auto carve = [&](size_t bytes) -> void* {
    void* p = (void*)w;
    w += (bytes + 255) & ~(size_t)255;
    return p;
  };
  __bf16* xb   = (__bf16*)carve((size_t)T_TOK * HDIM * 2);
  __bf16* hbuf = (__bf16*)carve((size_t)T_TOK * IDIM * 2);
  float* score  = (float*)carve((size_t)T_TOK * 4);
  float* gscore = (float*)carve((size_t)T_TOK * 4);
  int* eidv   = (int*)carve((size_t)T_TOK * 4);
  int* gperm  = (int*)carve((size_t)T_TOK * 4);
  int* idperm = (int*)carve((size_t)T_TOK * 4);
  int* counts = (int*)carve(NEXP * 4);
  int* cnt2   = (int*)carve(NEXP * 4);
  int* offs   = (int*)carve(NEXP * 4);
  int* sh_off = (int*)carve(4);
  int* sh_cnt = (int*)carve(4);

  // bf16 weight shadow copies (used only if scratch is big enough)
  const size_t SHW = (size_t)IDIM * HDIM;            // 2M elems per shared mat
  const size_t EXW = (size_t)NEXP * IDIM * HDIM;     // 32M elems per expert set
  __bf16* sgw_b = (__bf16*)carve(SHW * 2);
  __bf16* suw_b = (__bf16*)carve(SHW * 2);
  __bf16* sdw_b = (__bf16*)carve(SHW * 2);
  __bf16* egw_b = (__bf16*)carve(EXW * 2);
  __bf16* euw_b = (__bf16*)carve(EXW * 2);
  __bf16* edw_b = (__bf16*)carve(EXW * 2);
  const bool bf16w = ((size_t)(w - (uint8_t*)d_ws) <= ws_size);

  init_kernel<<<T_TOK / 256, 256, 0, stream>>>(counts, cnt2, sh_off, sh_cnt, idperm);
  cvtw_kernel<<<(T_TOK * HDIM) / 2048, 256, 0, stream>>>(x, xb);
  router_kernel<<<T_TOK / 8, 256, 0, stream>>>(x, rw, logits, score, eidv, counts);
  scan_kernel<<<1, 32, 0, stream>>>(counts, offs);
  scatter_kernel<<<T_TOK / 256, 256, 0, stream>>>(eidv, score, offs, cnt2, gperm, gscore);

  dim3 blk(256);
  const dim3 gu_sh(T_TOK / TILE_M, IDIM / TILE_N, 1);
  const dim3 dn_sh(T_TOK / TILE_M, HDIM / TILE_N, 1);
  const dim3 gu_ex(T_TOK / TILE_M, IDIM / TILE_N, NEXP);
  const dim3 dn_ex(T_TOK / TILE_M, HDIM / TILE_N, NEXP);

  if (bf16w) {
    cvtw_kernel<<<SHW / 2048, 256, 0, stream>>>(sgw, sgw_b);
    cvtw_kernel<<<SHW / 2048, 256, 0, stream>>>(suw, suw_b);
    cvtw_kernel<<<SHW / 2048, 256, 0, stream>>>(sdw, sdw_b);
    cvtw_kernel<<<EXW / 2048, 256, 0, stream>>>(egw, egw_b);
    cvtw_kernel<<<EXW / 2048, 256, 0, stream>>>(euw, euw_b);
    cvtw_kernel<<<EXW / 2048, 256, 0, stream>>>(edw, edw_b);

    gateup_kernel<__bf16><<<gu_sh, blk, 0, stream>>>(xb, sgw_b, suw_b, hbuf,
                                                     idperm, sh_off, sh_cnt);
    down_kernel<__bf16><<<dn_sh, blk, 0, stream>>>(hbuf, sdw_b, out, idperm,
                                                   nullptr, sh_off, sh_cnt, 0);
    gateup_kernel<__bf16><<<gu_ex, blk, 0, stream>>>(xb, egw_b, euw_b, hbuf,
                                                     gperm, offs, counts);
    down_kernel<__bf16><<<dn_ex, blk, 0, stream>>>(hbuf, edw_b, out, gperm,
                                                   gscore, offs, counts, 1);
  } else {
    gateup_kernel<float><<<gu_sh, blk, 0, stream>>>(xb, sgw, suw, hbuf,
                                                    idperm, sh_off, sh_cnt);
    down_kernel<float><<<dn_sh, blk, 0, stream>>>(hbuf, sdw, out, idperm,
                                                  nullptr, sh_off, sh_cnt, 0);
    gateup_kernel<float><<<gu_ex, blk, 0, stream>>>(xb, egw, euw, hbuf,
                                                    gperm, offs, counts);
    down_kernel<float><<<dn_ex, blk, 0, stream>>>(hbuf, edw, out, gperm,
                                                  gscore, offs, counts, 1);
  }
}
